// CytokineImputer_14851996909631
// MI455X (gfx1250) — matrix operand order repaired and checked by reading, not compiled, not run on hardware
//
#include <hip/hip_runtime.h>

// ---------------------------------------------------------------------------
// NNConv GNN fused for MI455X (gfx1250, wave32, WMMA).
//
// Reformulation: per-edge weight generation + message product is a GEMM
//   msg_tile(16x32) = A(16 x 32*CIN) @ B(32*CIN x 32)  (+ bias K-chunk)
// where A's K-chunk c is outer(x_e[c], h_e[:]) — built from one cached
// A-layout fragment of h scaled by a per-lane f16 scalar. B chunks are fixed
// 32x32 blocks of w2 (pre-transposed to f16), reused across all 160K edges
// (L1/L2 resident). Two edge-tiles per wave -> each B fragment feeds 4
// v_wmma_f32_16x16x32_f16. Aggregation via f32 global atomics (agg is 2.5MB,
// L2-resident). All other terms folded into small node-level kernels.
// ---------------------------------------------------------------------------

#define N_NODES 20000
#define N_EDGES 160000

typedef _Float16 half8 __attribute__((ext_vector_type(8)));
typedef _Float16 v16h  __attribute__((ext_vector_type(16)));
typedef float    v8f   __attribute__((ext_vector_type(8)));

__device__ __forceinline__ float silu_f(float v) { return v / (1.0f + __expf(-v)); }

// A-matrix (16x32 f16, MxK) fragment per ISA table:
// lane<16: M=lane, halves = K[8h..8h+7] ++ K[8h+16..8h+23] with h=0;
// lane>=16: same with h=1. 'row' points at 32 contiguous K halves for this M.
__device__ __forceinline__ v16h load_a_frag(const _Float16* row, int hi) {
  const half8* p = (const half8*)row;
  half8 lo = p[hi];        // K = 8*hi .. 8*hi+7
  half8 up = p[hi + 2];    // K = 8*hi+16 .. 8*hi+23
  return __builtin_shufflevector(lo, up, 0,1,2,3,4,5,6,7,8,9,10,11,12,13,14,15);
}

// B-matrix (32x16 f16, KxN) fragment: lane<16 -> N=lane, K=0..7 ++ 16..23;
// lane>=16 -> N=lane-16, K=8..15 ++ 24..31. 'col' points at 32 contiguous K halves
// for this N (w2 stored transposed).
__device__ __forceinline__ v16h load_b_frag(const _Float16* col, int hi) {
  const half8* p = (const half8*)col;
  return __builtin_shufflevector(p[hi], p[hi + 2], 0,1,2,3,4,5,6,7,8,9,10,11,12,13,14,15);
}

__device__ __forceinline__ v8f wmma_f16(v16h a, v16h b, v8f c) {
  // D = A x B + C, f32 accumulate
  return __builtin_amdgcn_wmma_f32_16x16x32_f16(false, a, false, b, (short)0, c,
                                                false, false);
}

// ---------------------------------------------------------------------------
// Weight preprocessing: transpose w2 -> f16 [col][m] so B fragments are
// contiguous 32B loads, and reshape b2 -> f16 [o][i] (layer1 padded to i=32).
// ---------------------------------------------------------------------------
__global__ void prep_weights_kernel(const float* __restrict__ w2_1,
                                    const float* __restrict__ b2_1,
                                    const float* __restrict__ w2_2,
                                    const float* __restrict__ b2_2,
                                    _Float16* __restrict__ w2T1,
                                    _Float16* __restrict__ b2T1,
                                    _Float16* __restrict__ w2T2,
                                    _Float16* __restrict__ b2T2) {
  int t0 = blockIdx.x * blockDim.x + threadIdx.x;
  int stride = blockDim.x * gridDim.x;
  for (int idx = t0; idx < 512 * 32; idx += stride) {
    int col = idx >> 5, m = idx & 31;
    w2T1[idx] = (_Float16)w2_1[m * 512 + col];
  }
  for (int idx = t0; idx < 1024 * 32; idx += stride) {
    int col = idx >> 5, m = idx & 31;
    w2T2[idx] = (_Float16)w2_2[m * 1024 + col];
  }
  for (int idx = t0; idx < 32 * 32; idx += stride) {
    int o = idx >> 5, i = idx & 31;
    b2T1[idx] = (_Float16)((i < 16) ? b2_1[i * 32 + o] : 0.0f);
    b2T2[idx] = (_Float16)b2_2[i * 32 + o];
  }
}

// ---------------------------------------------------------------------------
// In-degree count
// ---------------------------------------------------------------------------
__global__ void degree_kernel(const long long* __restrict__ ei, int* __restrict__ cnt) {
  int e = blockIdx.x * blockDim.x + threadIdx.x;
  if (e < N_EDGES) atomicAdd(&cnt[(int)ei[N_EDGES + e]], 1);
}

// ---------------------------------------------------------------------------
// Fused edge kernel. Each wave handles 2 tiles of 16 edges per iteration:
//   - stage src/dst, h_e = silu(ew*w1+b1) (f16), x[src] (f16, padded to 32)
//   - bias K-chunk: acc += X_tile @ B2
//   - K-chunks c=0..CIN-1: A = outer(x[:,c], h) via pk-f16 scale of cached
//     h fragment; B = w2T block; 4 WMMAs per B fragment (2 tiles x 2 N-halves)
//   - scatter: agg[dst] += D via global f32 atomics
// ---------------------------------------------------------------------------
template <int CIN>
__global__ __launch_bounds__(256) void edge_conv_kernel(
    const float* __restrict__ xin,         // N x CIN node features
    const long long* __restrict__ ei,      // 2 x E (src row then dst row)
    const float* __restrict__ ew,          // E
    const float* __restrict__ w1,          // 32
    const float* __restrict__ b1,          // 32
    const _Float16* __restrict__ w2T,      // (CIN*32) x 32, [col][m]
    const _Float16* __restrict__ b2T,      // 32 x 32, [o][i]
    float* __restrict__ agg,               // N x 32, pre-zeroed
    int npairs) {
  __shared__ __align__(16) _Float16 hstage[8][2][16][32];
  __shared__ __align__(16) _Float16 xstage[8][2][16][32];
  __shared__ int sstage[8][2][16];
  __shared__ int dstage[8][2][16];

  const int lane = threadIdx.x & 31;
  const int w    = threadIdx.x >> 5;
  const int hi   = lane >> 4;   // which half of the wave
  const int nl   = lane & 15;   // M index (A/C rows) and N index (B/C cols)
  const int gw   = blockIdx.x * 8 + w;
  const int nw   = gridDim.x * 8;

  for (int tp = gw; tp < npairs; tp += nw) {
    const int e0 = tp * 32;

    if (lane < 16) {
      sstage[w][0][lane] = (int)ei[e0 + lane];
      sstage[w][1][lane] = (int)ei[e0 + 16 + lane];
      dstage[w][0][lane] = (int)ei[N_EDGES + e0 + lane];
      dstage[w][1][lane] = (int)ei[N_EDGES + e0 + 16 + lane];
    }
    __builtin_amdgcn_wave_barrier();

    // h_e = silu(ew * w1 + b1): 32 edges x 32 values
    #pragma unroll 4
    for (int j = 0; j < 32; ++j) {
      int flat = j * 32 + lane;
      int u = flat >> 9, e = (flat >> 5) & 15, m = flat & 31;
      float v = fmaf(ew[e0 + u * 16 + e], w1[m], b1[m]);
      hstage[w][u][e][m] = (_Float16)silu_f(v);
    }
    // gather x[src], cast to f16, zero-pad K to 32
    #pragma unroll 4
    for (int j = 0; j < 32; ++j) {
      int flat = j * 32 + lane;
      int u = flat >> 9, e = (flat >> 5) & 15, i = flat & 31;
      float v = (i < CIN) ? xin[(size_t)sstage[w][u][e] * CIN + i] : 0.0f;
      xstage[w][u][e][i] = (_Float16)v;
    }
    __builtin_amdgcn_wave_barrier();

    v16h ah0 = load_a_frag(&hstage[w][0][nl][0], hi);
    v16h ah1 = load_a_frag(&hstage[w][1][nl][0], hi);
    v16h ax0 = load_a_frag(&xstage[w][0][nl][0], hi);
    v16h ax1 = load_a_frag(&xstage[w][1][nl][0], hi);

    v8f acc00 = {0.f,0.f,0.f,0.f,0.f,0.f,0.f,0.f};  // tile0, o in [0,16)
    v8f acc01 = {0.f,0.f,0.f,0.f,0.f,0.f,0.f,0.f};  // tile0, o in [16,32)
    v8f acc10 = {0.f,0.f,0.f,0.f,0.f,0.f,0.f,0.f};  // tile1, o in [0,16)
    v8f acc11 = {0.f,0.f,0.f,0.f,0.f,0.f,0.f,0.f};  // tile1, o in [16,32)

    { // bias K-chunk: C += X @ B2
      v16h bb0 = load_b_frag(b2T + (size_t)nl * 32, hi);
      v16h bb1 = load_b_frag(b2T + (size_t)(16 + nl) * 32, hi);
      acc00 = wmma_f16(ax0, bb0, acc00);
      acc10 = wmma_f16(ax1, bb0, acc10);
      acc01 = wmma_f16(ax0, bb1, acc01);
      acc11 = wmma_f16(ax1, bb1, acc11);
    }

    #pragma unroll 4
    for (int c = 0; c < CIN; ++c) {
      _Float16 xs0 = xstage[w][0][nl][c];
      _Float16 xs1 = xstage[w][1][nl][c];
      v16h a0 = ah0 * xs0;   // A chunk = outer(x[:,c], h)
      v16h a1 = ah1 * xs1;
      const _Float16* cb = w2T + (size_t)(c * 32 + nl) * 32;
      v16h bb0 = load_b_frag(cb, hi);
      v16h bb1 = load_b_frag(cb + 16 * 32, hi);
      acc00 = wmma_f16(a0, bb0, acc00);
      acc10 = wmma_f16(a1, bb0, acc10);
      acc01 = wmma_f16(a0, bb1, acc01);
      acc11 = wmma_f16(a1, bb1, acc11);
    }

    // scatter: C/D layout -> lane holds N=nl, rows M = r + 8*hi
    #pragma unroll
    for (int r = 0; r < 8; ++r) {
      int M  = r + 8 * hi;
      int d0 = dstage[w][0][M];
      int d1 = dstage[w][1][M];
      __hip_atomic_fetch_add(&agg[(size_t)d0 * 32 + nl],      acc00[r],
                             __ATOMIC_RELAXED, __HIP_MEMORY_SCOPE_AGENT);
      __hip_atomic_fetch_add(&agg[(size_t)d0 * 32 + 16 + nl], acc01[r],
                             __ATOMIC_RELAXED, __HIP_MEMORY_SCOPE_AGENT);
      __hip_atomic_fetch_add(&agg[(size_t)d1 * 32 + nl],      acc10[r],
                             __ATOMIC_RELAXED, __HIP_MEMORY_SCOPE_AGENT);
      __hip_atomic_fetch_add(&agg[(size_t)d1 * 32 + 16 + nl], acc11[r],
                             __ATOMIC_RELAXED, __HIP_MEMORY_SCOPE_AGENT);
    }
  }
}

// ---------------------------------------------------------------------------
// Node-level finalize: h1 = silu(agg1*inv_cnt + x@root1 + bias1)
// ---------------------------------------------------------------------------
__global__ void finalize1_kernel(const float* __restrict__ agg,
                                 const int* __restrict__ cnt,
                                 const float* __restrict__ x,
                                 const float* __restrict__ root,
                                 const float* __restrict__ bias,
                                 float* __restrict__ h1) {
  int idx = blockIdx.x * blockDim.x + threadIdx.x;
  if (idx >= N_NODES * 32) return;
  int n = idx >> 5, o = idx & 31;
  float inv = 1.0f / fmaxf((float)cnt[n], 1.0f);
  float v = fmaf(agg[idx], inv, bias[o]);
  #pragma unroll
  for (int i = 0; i < 16; ++i) v = fmaf(x[n * 16 + i], root[i * 32 + o], v);
  h1[idx] = silu_f(v);
}

// ---------------------------------------------------------------------------
// finalize2 fused with decode: out[n] = silu(agg2*inv + h1@root2 + bias2) @ dec_w + dec_b
// ---------------------------------------------------------------------------
__global__ void finalize2_kernel(const float* __restrict__ agg,
                                 const int* __restrict__ cnt,
                                 const float* __restrict__ h1,
                                 const float* __restrict__ root,
                                 const float* __restrict__ bias,
                                 const float* __restrict__ dw,
                                 const float* __restrict__ db,
                                 float* __restrict__ out) {
  int n = blockIdx.x * blockDim.x + threadIdx.x;
  if (n >= N_NODES) return;
  float inv = 1.0f / fmaxf((float)cnt[n], 1.0f);
  float res = db[0];
  #pragma unroll 4
  for (int o = 0; o < 32; ++o) {
    float v = fmaf(agg[n * 32 + o], inv, bias[o]);
    for (int i = 0; i < 32; ++i) v = fmaf(h1[n * 32 + i], root[i * 32 + o], v);
    res = fmaf(silu_f(v), dw[o], res);
  }
  out[n] = res;
}

// ---------------------------------------------------------------------------
extern "C" void kernel_launch(void* const* d_in, const int* in_sizes, int n_in,
                              void* d_out, int out_size, void* d_ws, size_t ws_size,
                              hipStream_t stream) {
  (void)in_sizes; (void)n_in; (void)out_size; (void)ws_size;
  const float*     x      = (const float*)d_in[0];
  const long long* ei     = (const long long*)d_in[1];
  const float*     ew     = (const float*)d_in[2];
  const float*     en1_w1 = (const float*)d_in[3];
  const float*     en1_b1 = (const float*)d_in[4];
  const float*     en1_w2 = (const float*)d_in[5];
  const float*     en1_b2 = (const float*)d_in[6];
  const float*     root1  = (const float*)d_in[7];
  const float*     bias1  = (const float*)d_in[8];
  const float*     en2_w1 = (const float*)d_in[9];
  const float*     en2_b1 = (const float*)d_in[10];
  const float*     en2_w2 = (const float*)d_in[11];
  const float*     en2_b2 = (const float*)d_in[12];
  const float*     root2  = (const float*)d_in[13];
  const float*     bias2  = (const float*)d_in[14];
  const float*     dec_w  = (const float*)d_in[15];
  const float*     dec_b  = (const float*)d_in[16];
  float* out = (float*)d_out;

  // workspace layout (256B-aligned slices)
  char* wsp = (char*)d_ws;
  size_t off = 0;
  auto take = [&](size_t bytes) -> void* {
    off = (off + 255) & ~(size_t)255;
    void* p = wsp + off;
    off += bytes;
    return p;
  };
  int*      cnt  = (int*)take((size_t)N_NODES * sizeof(int));
  float*    agg1 = (float*)take((size_t)N_NODES * 32 * sizeof(float));
  float*    agg2 = (float*)take((size_t)N_NODES * 32 * sizeof(float));
  float*    h1   = (float*)take((size_t)N_NODES * 32 * sizeof(float));
  _Float16* w2T1 = (_Float16*)take((size_t)512 * 32 * sizeof(_Float16));
  _Float16* b2T1 = (_Float16*)take((size_t)32 * 32 * sizeof(_Float16));
  _Float16* w2T2 = (_Float16*)take((size_t)1024 * 32 * sizeof(_Float16));
  _Float16* b2T2 = (_Float16*)take((size_t)32 * 32 * sizeof(_Float16));

  hipMemsetAsync(cnt,  0, (size_t)N_NODES * sizeof(int), stream);
  hipMemsetAsync(agg1, 0, (size_t)N_NODES * 32 * sizeof(float), stream);
  hipMemsetAsync(agg2, 0, (size_t)N_NODES * 32 * sizeof(float), stream);

  prep_weights_kernel<<<64, 256, 0, stream>>>(en1_w2, en1_b2, en2_w2, en2_b2,
                                              w2T1, b2T1, w2T2, b2T2);
  degree_kernel<<<(N_EDGES + 255) / 256, 256, 0, stream>>>(ei, cnt);

  const int npairs = N_EDGES / 32;  // 5000 wave-iterations of 32 edges
  edge_conv_kernel<16><<<625, 256, 0, stream>>>(x, ei, ew, en1_w1, en1_b1,
                                                w2T1, b2T1, agg1, npairs);
  finalize1_kernel<<<(N_NODES * 32 + 255) / 256, 256, 0, stream>>>(agg1, cnt, x,
                                                                   root1, bias1, h1);
  edge_conv_kernel<32><<<625, 256, 0, stream>>>(h1, ei, ew, en2_w1, en2_b1,
                                                w2T2, b2T2, agg2, npairs);
  finalize2_kernel<<<(N_NODES + 255) / 256, 256, 0, stream>>>(agg2, cnt, h1, root2,
                                                              bias2, dec_w, dec_b, out);
}
